// PointerGeneratorHead_352187318970
// MI455X (gfx1250) — matrix operand order repaired and checked by reading, not compiled, not run on hardware
//
#include <hip/hip_runtime.h>
#include <cstdint>
#include <cstddef>

#define DMODEL 1024
#define NB     8
#define NS     2048
#define NV     100279
#define NBOT   512
#define NHEADS 8
#define NHD    128
#define NTOK   64
#define TDIM   3072   // 3*DMODEL

typedef __attribute__((ext_vector_type(16))) __bf16      v16bf;
typedef __attribute__((ext_vector_type(8)))  float        v8f;
typedef __attribute__((ext_vector_type(8)))  unsigned int v8u;

union V8F { v8f v; float f[8]; };
union V8U { v8u v; uint4 q[2]; };
union U4B { uint4 u; __bf16 b[8]; };

__device__ __forceinline__ float sigmoidf(float x) { return 1.0f / (1.0f + __expf(-x)); }

// ---------------------------------------------------------------------------
// Generic small-M WMMA GEMM: out[chunk*16+m, n] = act( A[m,:]·W[n,:] + bias[n] )
//   A: fp32, chunked by blockIdx.y in 16-row tiles, Mrows valid rows per chunk
//   W: bf16 [N x K] row-major (K contiguous)  -> B operand
//   256 threads = 8 waves; each wave owns one 16-column tile.
//   Inner loop is software-pipelined (double-buffered k-slabs) so the
//   global/LDS loads for slab k+1 overlap the v_wmma for slab k.
//
// CDNA5 16-bit A-operand layout: lane l (row = l&15) holds K = lh*8..lh*8+7
// and K = 16+lh*8..16+lh*8+7 (lh = l>>4)  -> two contiguous uint4 LDS loads.
// B-operand: lane l (col = l&15) holds the same K as A -> two uint4 loads.
// ---------------------------------------------------------------------------
__global__ void __launch_bounds__(256) gemm16_kernel(
    const float* __restrict__ A, int Mrows,
    const __bf16* __restrict__ W, const float* __restrict__ bias,
    void* __restrict__ out, int outBf16, int N, int K, int act)
{
  __shared__ __align__(16) __bf16 As[16 * 1024];   // K <= 1024
  A += (size_t)blockIdx.y * 16 * (size_t)K;
  const int tid = threadIdx.x;
  const int tot = 16 * K;
  for (int idx = tid; idx < tot; idx += 256) {
    int m = idx / K;
    int k = idx - m * K;
    As[idx] = (m < Mrows) ? (__bf16)A[(size_t)m * K + k] : (__bf16)0.0f;
  }
  __syncthreads();

  const int wave = tid >> 5, lane = tid & 31;
  const int n0 = (blockIdx.x * 8 + wave) * 16;
  if (n0 >= N) return;
  const int cc  = lane & 15;
  const int lh8  = (lane >> 4) << 3;
  const int lh16 = (lane >> 4) << 4;
  int nrow = n0 + cc; if (nrow > N - 1) nrow = N - 1;   // clamp: in-bounds, masked at store
  const __bf16* __restrict__ wrow = W + (size_t)nrow * (size_t)K;

  V8U a_cur, b_cur, a_nxt, b_nxt;
  auto load_slab = [&](V8U& A8, V8U& B8, int kt) {
    const __bf16* ap = &As[cc * K + kt + lh8];
    A8.q[0] = *(const uint4*)(ap);          // ds_load_b128
    A8.q[1] = *(const uint4*)(ap + 16);     // ds_load_b128
    const __bf16* wp = wrow + kt + lh8;
    B8.q[0] = *(const uint4*)(wp);          // global_load_b128
    B8.q[1] = *(const uint4*)(wp + 16);     // global_load_b128
    __builtin_prefetch(wp + 128, 0, 3);     // near prefetch (WGP scope), 4 slabs ahead
  };

  v8f acc = {};
  load_slab(a_cur, b_cur, 0);
  for (int kt = 32; kt < K; kt += 32) {
    load_slab(a_nxt, b_nxt, kt);            // next slab in flight...
    acc = __builtin_amdgcn_wmma_f32_16x16x32_bf16(
        false, __builtin_bit_cast(v16bf, a_cur.v),
        false, __builtin_bit_cast(v16bf, b_cur.v),
        (short)0, acc, false, false);       // ...while this slab computes
    a_cur = a_nxt; b_cur = b_nxt;
  }
  acc = __builtin_amdgcn_wmma_f32_16x16x32_bf16(
      false, __builtin_bit_cast(v16bf, a_cur.v),
      false, __builtin_bit_cast(v16bf, b_cur.v),
      (short)0, acc, false, false);

  float bb = bias ? bias[nrow] : 0.0f;
  V8F r; r.v = acc;
  const int n = n0 + cc;
  if (n >= N) return;
  const int mhi = (lane >> 4) << 3;
#pragma unroll
  for (int rr = 0; rr < 8; ++rr) {
    int m = rr + mhi;
    if (m >= Mrows) continue;
    float vl = r.f[rr] + bb;
    if (act == 1) vl = 0.5f * vl * (1.0f + erff(vl * 0.7071067811865476f));  // exact gelu
    size_t o = ((size_t)blockIdx.y * 16 + (size_t)m) * (size_t)N + (size_t)n;
    if (outBf16) ((__bf16*)out)[o] = (__bf16)vl;
    else         ((float*)out)[o]  = vl;
  }
}

// ---------------------------------------------------------------------------
__global__ void __launch_bounds__(256) f32_to_bf16_kernel(
    const float* __restrict__ src, __bf16* __restrict__ dst, int n) {
  int i = blockIdx.x * 256 + threadIdx.x;
  if (i < n) dst[i] = (__bf16)src[i];
}

__global__ void __launch_bounds__(256) copy_f32_kernel(
    const float* __restrict__ src, float* __restrict__ dst, int n) {
  int i = blockIdx.x * 256 + threadIdx.x;
  if (i < n) dst[i] = src[i];
}

__global__ void __launch_bounds__(256) zero_f32_kernel(float* __restrict__ p, int n) {
  int i = blockIdx.x * 256 + threadIdx.x;
  if (i < n) p[i] = 0.0f;
}

// ---------------------------------------------------------------------------
// Attention: one block per (b, head). Scores + softmax + weighted V sum.
// ---------------------------------------------------------------------------
__global__ void __launch_bounds__(256) attention_kernel(
    const float* __restrict__ q, const __bf16* __restrict__ kbf,
    const __bf16* __restrict__ vbf, float* __restrict__ attn,
    float* __restrict__ ctx)
{
  __shared__ float sc[NS];
  __shared__ float qs[NHD];
  __shared__ float red[256];
  const int tid = threadIdx.x;
  const int b = blockIdx.x / NHEADS;
  const int h = blockIdx.x - b * NHEADS;

  if (tid < NHD) qs[tid] = q[b * DMODEL + h * NHD + tid];
  __syncthreads();

  const float scale = 0.08838834764831845f;  // 1/sqrt(128)
  for (int s = tid; s < NS; s += 256) {
    const uint4* kp = (const uint4*)(kbf + ((size_t)(b * NS + s)) * DMODEL + h * NHD);
    float acc = 0.0f;
#pragma unroll
    for (int j = 0; j < 16; ++j) {
      U4B t; t.u = kp[j];
#pragma unroll
      for (int e = 0; e < 8; ++e) acc += qs[j * 8 + e] * (float)t.b[e];
    }
    sc[s] = acc * scale;
  }
  __syncthreads();

  // max
  float m = -1e30f;
  for (int s = tid; s < NS; s += 256) m = fmaxf(m, sc[s]);
  red[tid] = m; __syncthreads();
  for (int o = 128; o > 0; o >>= 1) { if (tid < o) red[tid] = fmaxf(red[tid], red[tid + o]); __syncthreads(); }
  const float mx = red[0]; __syncthreads();
  // exp + sum
  float sum = 0.0f;
  for (int s = tid; s < NS; s += 256) { float e = __expf(sc[s] - mx); sc[s] = e; sum += e; }
  red[tid] = sum; __syncthreads();
  for (int o = 128; o > 0; o >>= 1) { if (tid < o) red[tid] += red[tid + o]; __syncthreads(); }
  const float inv = 1.0f / red[0]; __syncthreads();
  for (int s = tid; s < NS; s += 256) {
    float p = sc[s] * inv; sc[s] = p;
    attn[((size_t)(b * NHEADS + h)) * NS + s] = p;
  }
  __syncthreads();

  // ctx[d] = sum_s p[s] * V[b,s,h,d]; 128 threads own one d each (coalesced in d)
  if (tid < NHD) {
    float acc = 0.0f;
    for (int s = 0; s < NS; ++s)
      acc += sc[s] * (float)vbf[((size_t)(b * NS + s)) * DMODEL + h * NHD + tid];
    ctx[b * DMODEL + h * NHD + tid] = acc;
  }
}

// ---------------------------------------------------------------------------
__global__ void __launch_bounds__(256) scatter_copy_kernel(
    const float* __restrict__ attn, const int* __restrict__ ids,
    float* __restrict__ copyd)
{
  int idx = blockIdx.x * 256 + threadIdx.x;   // over B*S
  if (idx >= NB * NS) return;
  int b = idx / NS, s = idx - b * NS;
  float w = 0.0f;
#pragma unroll
  for (int h = 0; h < NHEADS; ++h) w += attn[((size_t)(b * NHEADS + h)) * NS + s];
  w *= (1.0f / NHEADS);
  int id = ids[idx];
  if (id >= 0) atomicAdd(&copyd[(size_t)b * NV + id], w);
}

__global__ void __launch_bounds__(256) pgen_kernel(
    const float* __restrict__ h, const float* __restrict__ ctxp,
    const float* __restrict__ prev, const float* __restrict__ w,
    const float* __restrict__ bptr, float* __restrict__ pg)
{
  __shared__ float red[256];
  const int b = blockIdx.x, tid = threadIdx.x;
  float acc = 0.0f;
  for (int j = tid; j < TDIM; j += 256) {
    float x = (j < DMODEL) ? h[b * DMODEL + j]
            : (j < 2 * DMODEL) ? ctxp[b * DMODEL + (j - DMODEL)]
            : prev[b * DMODEL + (j - 2 * DMODEL)];
    acc += x * w[j];
  }
  red[tid] = acc; __syncthreads();
  for (int o = 128; o > 0; o >>= 1) { if (tid < o) red[tid] += red[tid + o]; __syncthreads(); }
  if (tid == 0) pg[b] = sigmoidf(red[0] + bptr[0]);
}

__global__ void __launch_bounds__(256) rowmax_kernel(
    const float* __restrict__ logits, float* __restrict__ mx)
{
  __shared__ float red[256];
  const int b = blockIdx.x, tid = threadIdx.x;
  float m = -1e30f;
  for (int v = tid; v < NV; v += 256) m = fmaxf(m, logits[(size_t)b * NV + v]);
  red[tid] = m; __syncthreads();
  for (int o = 128; o > 0; o >>= 1) { if (tid < o) red[tid] = fmaxf(red[tid], red[tid + o]); __syncthreads(); }
  if (tid == 0) mx[b] = red[0];
}

__global__ void __launch_bounds__(256) rowsumexp_kernel(
    const float* __restrict__ logits, const float* __restrict__ mx,
    float* __restrict__ sm)
{
  __shared__ float red[256];
  const int b = blockIdx.x, tid = threadIdx.x;
  const float m = mx[b];
  float s = 0.0f;
  for (int v = tid; v < NV; v += 256) s += __expf(logits[(size_t)b * NV + v] - m);
  red[tid] = s; __syncthreads();
  for (int o = 128; o > 0; o >>= 1) { if (tid < o) red[tid] += red[tid + o]; __syncthreads(); }
  if (tid == 0) sm[b] = red[0];
}

__global__ void __launch_bounds__(256) argmax_kernel(
    const float* __restrict__ logits, const float* __restrict__ mx,
    const float* __restrict__ sm, const float* __restrict__ copyd,
    const float* __restrict__ pg, int* __restrict__ tokens, int t)
{
  __shared__ float rv[256];
  __shared__ int   ri[256];
  const int b = blockIdx.x, tid = threadIdx.x;
  const float p = pg[b], m = mx[b], inv = 1.0f / sm[b], q1 = 1.0f - p;
  float bestv = -1e30f; int besti = 0;
  for (int v = tid; v < NV; v += 256) {
    float f = p * __expf(logits[(size_t)b * NV + v] - m) * inv
            + q1 * copyd[(size_t)b * NV + v];
    if (f > bestv) { bestv = f; besti = v; }
  }
  rv[tid] = bestv; ri[tid] = besti; __syncthreads();
  for (int o = 128; o > 0; o >>= 1) {
    if (tid < o) {
      if (rv[tid + o] > rv[tid] || (rv[tid + o] == rv[tid] && ri[tid + o] < ri[tid])) {
        rv[tid] = rv[tid + o]; ri[tid] = ri[tid + o];
      }
    }
    __syncthreads();
  }
  if (tid == 0) tokens[b * NTOK + t] = ri[0];
}

__global__ void __launch_bounds__(256) gru_kernel(
    const float* __restrict__ gi, const float* __restrict__ gh,
    const float* __restrict__ gc, float* __restrict__ h,
    float* __restrict__ prev, const float* __restrict__ ctxp)
{
  int i = blockIdx.x * 256 + threadIdx.x;
  if (i >= NB * DMODEL) return;
  int b = i / DMODEL, d = i - b * DMODEL;
  const float* gib = gi + (size_t)b * TDIM;
  const float* ghb = gh + (size_t)b * TDIM;
  const float* gcb = gc + (size_t)b * TDIM;
  float r = sigmoidf(gib[d]              + ghb[d]              + gcb[d]);
  float z = sigmoidf(gib[DMODEL + d]     + ghb[DMODEL + d]     + gcb[DMODEL + d]);
  float n = tanhf  (gib[2 * DMODEL + d]  + gcb[2 * DMODEL + d] + r * ghb[2 * DMODEL + d]);
  float hn = (1.0f - z) * n + z * h[i];
  h[i]    = hn;
  prev[i] = ctxp[i];   // next step's prev_embed = this step's attention output
}

// ---------------------------------------------------------------------------
extern "C" void kernel_launch(void* const* d_in, const int* in_sizes, int n_in,
                              void* d_out, int out_size, void* d_ws, size_t ws_size,
                              hipStream_t stream)
{
  (void)in_sizes; (void)n_in; (void)out_size; (void)ws_size;
  const float* hidden   = (const float*)d_in[0];
  const float* context  = (const float*)d_in[1];
  const int*   ids      = (const int*)d_in[2];
  /* d_in[3] = max_tokens scalar; fixed NTOK=64 at compile time */
  const float* down_w   = (const float*)d_in[4];
  const float* down_b   = (const float*)d_in[5];
  const float* up_w     = (const float*)d_in[6];
  const float* inproj_w = (const float*)d_in[7];
  const float* inproj_b = (const float*)d_in[8];
  const float* outproj_w= (const float*)d_in[9];
  const float* outproj_b= (const float*)d_in[10];
  const float* pg_w     = (const float*)d_in[11];
  const float* pg_b     = (const float*)d_in[12];
  const float* gih_w    = (const float*)d_in[13];
  const float* ghh_w    = (const float*)d_in[14];
  const float* gch_w    = (const float*)d_in[15];
  const float* gih_b    = (const float*)d_in[16];
  const float* ghh_b    = (const float*)d_in[17];
  const float* iproj_w  = (const float*)d_in[18];
  const float* iproj_b  = (const float*)d_in[19];
  int* tokens = (int*)d_out;

  char* ws = (char*)d_ws;
  size_t off = 0;
  auto carve = [&](size_t bytes) -> void* {
    void* p = ws + off;
    off = (off + bytes + 255) & ~(size_t)255;
    return p;
  };

  // bf16 weight copies (one-time; L2-resident working set afterwards)
  __bf16* inproj_bf  = (__bf16*)carve((size_t)TDIM * DMODEL * 2);
  __bf16* outproj_bf = (__bf16*)carve((size_t)DMODEL * DMODEL * 2);
  __bf16* down_bf    = (__bf16*)carve((size_t)NBOT * DMODEL * 2);
  __bf16* up_bf      = (__bf16*)carve((size_t)NV * NBOT * 2);
  __bf16* gih_bf     = (__bf16*)carve((size_t)TDIM * DMODEL * 2);
  __bf16* ghh_bf     = (__bf16*)carve((size_t)TDIM * DMODEL * 2);
  __bf16* gch_bf     = (__bf16*)carve((size_t)TDIM * DMODEL * 2);
  __bf16* iproj_bf   = (__bf16*)carve((size_t)DMODEL * DMODEL * 2);
  // bf16 K/V caches (hoisted out of the decode loop)
  __bf16* k_bf       = (__bf16*)carve((size_t)NB * NS * DMODEL * 2);
  __bf16* v_bf       = (__bf16*)carve((size_t)NB * NS * DMODEL * 2);
  // fp32 state / scratch
  float* hbuf   = (float*)carve((size_t)NB * DMODEL * 4);
  float* prev   = (float*)carve((size_t)NB * DMODEL * 4);
  float* qbuf   = (float*)carve((size_t)NB * DMODEL * 4);
  float* ctxbuf = (float*)carve((size_t)NB * DMODEL * 4);
  float* ctxp   = (float*)carve((size_t)NB * DMODEL * 4);
  float* hb     = (float*)carve((size_t)NB * NBOT * 4);
  float* attn   = (float*)carve((size_t)NB * NHEADS * NS * 4);
  float* logits = (float*)carve((size_t)NB * NV * 4);
  float* copyd  = (float*)carve((size_t)NB * NV * 4);
  float* giB    = (float*)carve((size_t)NB * TDIM * 4);
  float* ghB    = (float*)carve((size_t)NB * TDIM * 4);
  float* gcB    = (float*)carve((size_t)NB * TDIM * 4);
  float* pgv    = (float*)carve(256);
  float* mxb    = (float*)carve(256);
  float* smb    = (float*)carve(256);

  auto conv = [&](const float* s, __bf16* d, size_t n) {
    f32_to_bf16_kernel<<<dim3((unsigned)((n + 255) / 256)), 256, 0, stream>>>(s, d, (int)n);
  };

  // ---- Phase 0: weight conversion, prev0, K/V projection ----
  conv(inproj_w,  inproj_bf,  (size_t)TDIM * DMODEL);
  conv(outproj_w, outproj_bf, (size_t)DMODEL * DMODEL);
  conv(down_w,    down_bf,    (size_t)NBOT * DMODEL);
  conv(up_w,      up_bf,      (size_t)NV * NBOT);
  conv(gih_w,     gih_bf,     (size_t)TDIM * DMODEL);
  conv(ghh_w,     ghh_bf,     (size_t)TDIM * DMODEL);
  conv(gch_w,     gch_bf,     (size_t)TDIM * DMODEL);
  conv(iproj_w,   iproj_bf,   (size_t)DMODEL * DMODEL);

  copy_f32_kernel<<<dim3(32), 256, 0, stream>>>(hidden, hbuf, NB * DMODEL);
  // prev0 = hidden @ input_proj^T + b
  gemm16_kernel<<<dim3(8, 1), 256, 0, stream>>>(hidden, NB, iproj_bf, iproj_b,
                                                prev, 0, DMODEL, DMODEL, 0);
  // K = context @ wk^T + bk ; V = context @ wv^T + bv  (M = B*S = 16384 in 16-row chunks)
  gemm16_kernel<<<dim3(8, 1024), 256, 0, stream>>>(context, 16,
      inproj_bf + (size_t)DMODEL * DMODEL, inproj_b + DMODEL,
      k_bf, 1, DMODEL, DMODEL, 0);
  gemm16_kernel<<<dim3(8, 1024), 256, 0, stream>>>(context, 16,
      inproj_bf + (size_t)2 * DMODEL * DMODEL, inproj_b + 2 * DMODEL,
      v_bf, 1, DMODEL, DMODEL, 0);

  // ---- Phase 1: 64 sequential decode steps ----
  const unsigned vgrid = (unsigned)(((NV + 15) / 16 + 7) / 8);          // 784
  const unsigned zgrid = (unsigned)((NB * NV + 255) / 256);             // zero copy-dist
  for (int t = 0; t < NTOK; ++t) {
    // q = h @ wq^T + bq
    gemm16_kernel<<<dim3(8, 1), 256, 0, stream>>>(hbuf, NB, inproj_bf, inproj_b,
                                                  qbuf, 0, DMODEL, DMODEL, 0);
    attention_kernel<<<dim3(NB * NHEADS), 256, 0, stream>>>(qbuf, k_bf, v_bf, attn, ctxbuf);
    // ctx = attn_out @ out_proj^T + b
    gemm16_kernel<<<dim3(8, 1), 256, 0, stream>>>(ctxbuf, NB, outproj_bf, outproj_b,
                                                  ctxp, 0, DMODEL, DMODEL, 0);
    // hb = gelu(h @ vocab_down^T + b)
    gemm16_kernel<<<dim3(4, 1), 256, 0, stream>>>(hbuf, NB, down_bf, down_b,
                                                  hb, 0, NBOT, DMODEL, 1);
    // logits = hb @ vocab_up^T   (the perf-critical 102 MB bf16 stream, L2-resident)
    gemm16_kernel<<<dim3(vgrid, 1), 256, 0, stream>>>(hb, NB, up_bf, nullptr,
                                                      logits, 0, NV, NBOT, 0);
    zero_f32_kernel<<<dim3(zgrid), 256, 0, stream>>>(copyd, NB * NV);
    scatter_copy_kernel<<<dim3(64), 256, 0, stream>>>(attn, ids, copyd);
    pgen_kernel<<<dim3(NB), 256, 0, stream>>>(hbuf, ctxp, prev, pg_w, pg_b, pgv);
    rowmax_kernel<<<dim3(NB), 256, 0, stream>>>(logits, mxb);
    rowsumexp_kernel<<<dim3(NB), 256, 0, stream>>>(logits, mxb, smb);
    argmax_kernel<<<dim3(NB), 256, 0, stream>>>(logits, mxb, smb, copyd, pgv, tokens, t);
    // GRU gates
    gemm16_kernel<<<dim3(24, 1), 256, 0, stream>>>(ctxp, NB, gih_bf, gih_b,
                                                   giB, 0, TDIM, DMODEL, 0);
    gemm16_kernel<<<dim3(24, 1), 256, 0, stream>>>(hbuf, NB, ghh_bf, ghh_b,
                                                   ghB, 0, TDIM, DMODEL, 0);
    gemm16_kernel<<<dim3(24, 1), 256, 0, stream>>>(ctxp, NB, gch_bf, nullptr,
                                                   gcB, 0, TDIM, DMODEL, 0);
    gru_kernel<<<dim3(32), 256, 0, stream>>>(giB, ghB, gcB, hbuf, prev, ctxp);
  }
}
